// GPT_56865366999135
// MI455X (gfx1250) — compile-verified, matrix operand order checked by reading, not run on hardware
//
#include <hip/hip_runtime.h>
#include <hip/hip_bf16.h>
#include <math.h>

// GPT forward (B=8, L=512, E=512, H=8, HD=64, DEPTH=8, FFN=4E) for MI455X gfx1250.
// Matrix-throughput bound (~240 GFLOPs, weights L2-resident) -> all GEMMs and
// attention matmuls run on v_wmma_f32_16x16x32_f16, wave32, 32x64 tile/wave.
// Weights are pre-transposed to f16 [N,K] per layer so BOTH operand fragments
// load as contiguous K-pairs (compiler emits global_load_b128, no converts).

#define LSEQ 512
#define EDIM 512
#define NHEAD 8
#define HD 64
#define DEPTH 8
#define NTOK 4096   // B*L
#define FFN 2048    // 4*E

typedef __attribute__((ext_vector_type(16))) _Float16 v16h;
typedef __attribute__((ext_vector_type(8)))  float    v8f;

// ---------------------------------------------------------------- WMMA helpers
// K index for half i (0..15) of an A/B fragment, per CDNA5 ISA 7.12.2:
//   VGPR p = i>>1 ; p<4 -> K base 0, p>=4 -> K base 16 ; lanes>=16 add 8.
__device__ __forceinline__ int kmap(int i, int hi) {
  int p = i >> 1;
  return ((p & 4) << 2) + (hi << 3) + ((p & 3) << 1) + (i & 1);
}

// Fragment with per-lane row (lane&15) and contiguous K pairs, row-major, ld.
// Used for: A tiles, K^T tiles (K row-major == B[k][n] gather), and
// transposed-f16 weight tiles Bt[n*K+k].
__device__ __forceinline__ v16h load_frag(const _Float16* __restrict__ base,
                                          int ld, int lane) {
  int row = lane & 15, hi = (lane >> 4) & 1;
  v16h a;
#pragma unroll
  for (int i = 0; i < 16; ++i) a[i] = base[row * ld + kmap(i, hi)];
  return a;
}

__device__ __forceinline__ v8f wmma32(v16h a, v16h b, v8f c) {
  return __builtin_amdgcn_wmma_f32_16x16x32_f16(false, a, false, b,
                                                (short)0, c, false, false);
}

__device__ __forceinline__ float gelu_tanh(float x) {
  float x3 = x * x * x;
  return 0.5f * x * (1.0f + tanhf(0.7978845608028654f * (x + 0.044715f * x3)));
}

// --------------------------------------------------------- embedding + pos-enc
__global__ __launch_bounds__(256) void embed_kernel(
    const int* __restrict__ s, const float* __restrict__ embed,
    float* __restrict__ xf, _Float16* __restrict__ xh) {
  int bl = blockIdx.x;            // b*L + l
  int b  = bl >> 9;
  int l  = bl & (LSEQ - 1);
  int tok = (l == 0) ? 0 : s[b * LSEQ + l - 1];   // roll-right, start token 0
  // pe_T[l][e] = (l even) ? sin(e*w) : cos(e*w),  w = exp(-(l&~1)*ln(1e4)/L)
  float w = __expf(-(float)(l & ~1) * (9.210340371976184f / (float)LSEQ));
  for (int e = threadIdx.x; e < EDIM; e += 256) {
    float ang = (float)e * w;
    float pe  = (l & 1) ? __cosf(ang) : __sinf(ang);
    float v   = embed[tok * EDIM + e] + pe;
    xf[bl * EDIM + e] = v;
    xh[bl * EDIM + e] = (_Float16)v;
  }
}

// ------------------------------------------- per-layer weight transpose -> f16
// Converts one layer's {wq,wk,wv,wo,w1,w2} (f32 [K,N]) into f16 [N,K] packed
// into a single 3,145,728-element buffer. Coalesced writes; reads hit L2.
#define WT_WQ 0
#define WT_WK 262144
#define WT_WV 524288
#define WT_WO 786432
#define WT_W1 1048576
#define WT_W2 2097152
#define WT_TOTAL 3145728

__global__ __launch_bounds__(256) void wconv_kernel(
    const float* __restrict__ wq, const float* __restrict__ wk,
    const float* __restrict__ wv, const float* __restrict__ wo,
    const float* __restrict__ w1, const float* __restrict__ w2,
    _Float16* __restrict__ out) {
  int idx = blockIdx.x * 256 + threadIdx.x;
  float v;
  if (idx < WT_W1) {                       // wq/wk/wv/wo: K=512, N=512
    int t = idx >> 18;
    int r = idx & 262143;
    int n = r >> 9, k = r & 511;
    const float* w = (t == 0) ? wq : (t == 1) ? wk : (t == 2) ? wv : wo;
    v = w[k * 512 + n];
  } else if (idx < WT_W2) {                // w1: K=512, N=2048
    int r = idx - WT_W1;
    int n = r >> 9, k = r & 511;
    v = w1[k * FFN + n];
  } else {                                 // w2: K=2048, N=512
    int r = idx - WT_W2;
    int n = r >> 11, k = r & 2047;
    v = w2[k * EDIM + n];
  }
  out[idx] = (_Float16)v;
}

// ------------------------------------------------------------------- WMMA GEMM
// C[M,N] = A(f16,[M,K]) x Bt(f16,[N,K]) + bias ; ACT=1 -> GELU ; RES -> +resid
// and write f32 master copy ; TRANSOUT -> scatter f16 out as vT[b][h][hd][l].
// One wave computes a 32x64 strip (2 A-frags x 4 B-frags, 8 WMMAs / K-step).
template <int ACT, bool RES, bool TRANSOUT>
__global__ __launch_bounds__(256) void gemm_kernel(
    const _Float16* __restrict__ A, const _Float16* __restrict__ Bt,
    const float* __restrict__ bias, const float* __restrict__ resid,
    float* __restrict__ outF, _Float16* __restrict__ outH,
    int M, int N, int K) {
  int lane = threadIdx.x & 31;
  int wid  = blockIdx.x * 8 + (threadIdx.x >> 5);
  int numNB = N >> 6;                       // 64-wide column blocks
  int mt = wid / numNB, nb = wid % numNB;
  if (mt >= (M >> 5)) return;               // uniform per wave
  int m0 = mt << 5, n0 = nb << 6;

  v8f acc[2][4] = {};
  for (int kk = 0; kk < K; kk += 32) {
    v16h a0 = load_frag(A + m0 * K + kk, K, lane);
    v16h a1 = load_frag(A + (m0 + 16) * K + kk, K, lane);
#pragma unroll
    for (int t = 0; t < 4; ++t) {
      v16h b = load_frag(Bt + (n0 + t * 16) * K + kk, K, lane);
      acc[0][t] = wmma32(a0, b, acc[0][t]);
      acc[1][t] = wmma32(a1, b, acc[1][t]);
    }
  }

  int col = lane & 15, hi = (lane >> 4) & 1;
#pragma unroll
  for (int half = 0; half < 2; ++half) {
#pragma unroll
    for (int t = 0; t < 4; ++t) {
#pragma unroll
      for (int r = 0; r < 8; ++r) {
        int m = m0 + half * 16 + r + (hi << 3);  // C/D: M = vgpr + 8*laneHi
        int n = n0 + t * 16 + col;
        float v = acc[half][t][r] + bias[n];
        if (ACT == 1) v = gelu_tanh(v);
        if (RES) {
          v += resid[m * N + n];
          outF[m * N + n] = v;
        }
        if (TRANSOUT) {                    // vT[((b*H + h)*HD + hd)*L + l]
          int bb = m >> 9, l = m & (LSEQ - 1);
          int hh = n >> 6, hd = n & (HD - 1);
          outH[((bb * NHEAD + hh) * HD + hd) * LSEQ + l] = (_Float16)v;
        } else {
          outH[m * N + n] = (_Float16)v;
        }
      }
    }
  }
}

// ------------------------------------------------------------ causal attention
// One wave per (qt, h, b): 16 query rows. Scores via WMMA into LDS, per-row
// softmax bounded by the causal limit, ctx = P x V^T via WMMA (P from LDS).
__global__ __launch_bounds__(32) void attn_kernel(
    const _Float16* __restrict__ qh, const _Float16* __restrict__ kh,
    const _Float16* __restrict__ vT, _Float16* __restrict__ ctxh) {
  __shared__ float    S[16][LSEQ];
  __shared__ _Float16 P[16][LSEQ];

  int lane = threadIdx.x;
  int qt = blockIdx.x, h = blockIdx.y, b = blockIdx.z;
  int q0 = qt << 4;
  int base = (b * LSEQ) * EDIM + h * HD;    // + l*EDIM + hd   (q/k/ctx layout)
  int vbase = (b * NHEAD + h) * HD * LSEQ;  // + hd*LSEQ + l   (vT layout)
  int col = lane & 15, hi = (lane >> 4) & 1;
  const float scale = 0.125f;               // 1/sqrt(64)

  // scores: only tiles mt <= qt ever contribute under the causal mask
  for (int mt = 0; mt <= qt; ++mt) {
    v8f sc = {};
#pragma unroll
    for (int kk = 0; kk < HD; kk += 32) {
      v16h a  = load_frag(qh + base + q0 * EDIM + kk, EDIM, lane);
      // K^T as B: B[k][n] = K[key=n][k]; row-major K gather == frag pattern
      v16h kb = load_frag(kh + base + (mt << 4) * EDIM + kk, EDIM, lane);
      sc = wmma32(a, kb, sc);
    }
#pragma unroll
    for (int r = 0; r < 8; ++r)
      S[r + (hi << 3)][(mt << 4) + col] = sc[r] * scale;
  }
  __syncthreads();

  // per-row softmax, causal limit inclusive; P zero-filled beyond limit
  if (lane < 16) {
    int limit = q0 + lane;
    float mx = -1e30f;
    for (int m = 0; m <= limit; ++m) mx = fmaxf(mx, S[lane][m]);
    float sum = 0.f;
    for (int m = 0; m <= limit; ++m) {
      float e = __expf(S[lane][m] - mx);
      sum += e;
      S[lane][m] = e;
    }
    float inv = 1.f / sum;
    for (int m = 0; m <= limit; ++m) P[lane][m] = (_Float16)(S[lane][m] * inv);
    for (int m = limit + 1; m < LSEQ; ++m) P[lane][m] = (_Float16)0.f;
  }
  __syncthreads();

  // ctx[16,64] = P[16,512] x V[512,64];  B[k][n] = V[key=k][hd=n] = vT[n][k]
  v8f acc[4] = {};
  for (int kk = 0; kk < LSEQ; kk += 32) {
    v16h a = load_frag(&P[0][0] + kk, LSEQ, lane);
#pragma unroll
    for (int t = 0; t < 4; ++t) {
      v16h vb = load_frag(vT + vbase + (t * 16) * LSEQ + kk, LSEQ, lane);
      acc[t] = wmma32(a, vb, acc[t]);
    }
  }
#pragma unroll
  for (int t = 0; t < 4; ++t)
#pragma unroll
    for (int r = 0; r < 8; ++r)
      ctxh[base + (q0 + r + (hi << 3)) * EDIM + t * 16 + col] =
          (_Float16)acc[t][r];
}

// -------------------------------------- final projection + log_softmax + gather
__global__ __launch_bounds__(256) void final_kernel(
    const float* __restrict__ xf, const float* __restrict__ wf,
    const float* __restrict__ bf, const int* __restrict__ s,
    float* __restrict__ out) {
  __shared__ float red[256];
  int b = blockIdx.x;
  float total = 0.f;
  for (int l = threadIdx.x; l < LSEQ; l += 256) {
    const float* xp = xf + (b * LSEQ + l) * EDIM;
    float a0 = bf[0], a1 = bf[1];
    for (int e = 0; e < EDIM; ++e) {
      float xv = xp[e];
      a0 += xv * wf[2 * e];
      a1 += xv * wf[2 * e + 1];
    }
    float mx  = fmaxf(a0, a1);
    float lse = mx + __logf(__expf(a0 - mx) + __expf(a1 - mx));
    int sv = s[b * LSEQ + l];
    total += ((sv == 0) ? a0 : a1) - lse;
  }
  red[threadIdx.x] = total;
  __syncthreads();
  for (int st = 128; st > 0; st >>= 1) {
    if (threadIdx.x < st) red[threadIdx.x] += red[threadIdx.x + st];
    __syncthreads();
  }
  if (threadIdx.x == 0) out[b] = red[0] * 0.5f;
}

// ----------------------------------------------------------------------- driver
extern "C" void kernel_launch(void* const* d_in, const int* in_sizes, int n_in,
                              void* d_out, int out_size, void* d_ws, size_t ws_size,
                              hipStream_t stream) {
  (void)in_sizes; (void)n_in; (void)out_size; (void)ws_size;
  const int*   s     = (const int*)  d_in[0];
  const float* embed = (const float*)d_in[1];
  const float* wq    = (const float*)d_in[2];
  const float* bq    = (const float*)d_in[3];
  const float* wk    = (const float*)d_in[4];
  const float* bk    = (const float*)d_in[5];
  const float* wv    = (const float*)d_in[6];
  const float* bv    = (const float*)d_in[7];
  const float* wo    = (const float*)d_in[8];
  const float* bo    = (const float*)d_in[9];
  const float* w1    = (const float*)d_in[10];
  const float* b1    = (const float*)d_in[11];
  const float* w2    = (const float*)d_in[12];
  const float* b2    = (const float*)d_in[13];
  const float* wf    = (const float*)d_in[14];
  const float* bf    = (const float*)d_in[15];
  float* out = (float*)d_out;

  // workspace carve (~50 MB)
  char* p = (char*)d_ws;
  float*     xf   = (float*)p;     p += (size_t)NTOK * EDIM * 4;  // 8 MB
  _Float16*  xh   = (_Float16*)p;  p += (size_t)NTOK * EDIM * 2;  // 4 MB
  _Float16*  qh   = (_Float16*)p;  p += (size_t)NTOK * EDIM * 2;
  _Float16*  kh   = (_Float16*)p;  p += (size_t)NTOK * EDIM * 2;
  _Float16*  vT   = (_Float16*)p;  p += (size_t)NTOK * EDIM * 2;
  _Float16*  ctxh = (_Float16*)p;  p += (size_t)NTOK * EDIM * 2;
  _Float16*  hbuf = (_Float16*)p;  p += (size_t)NTOK * FFN  * 2;  // 16 MB
  _Float16*  wt   = (_Float16*)p;  p += (size_t)WT_TOTAL * 2;     // 6.3 MB

  embed_kernel<<<NTOK, 256, 0, stream>>>(s, embed, xf, xh);

  const int blkE  = (NTOK / 32) * (EDIM / 64) / 8;   // 128
  const int blkF1 = (NTOK / 32) * (FFN  / 64) / 8;   // 512

  for (int d = 0; d < DEPTH; ++d) {
    // per-layer f16 [N,K] weight transposes (stream-ordered, buffer reused)
    wconv_kernel<<<WT_TOTAL / 256, 256, 0, stream>>>(
        wq + (size_t)d * EDIM * EDIM, wk + (size_t)d * EDIM * EDIM,
        wv + (size_t)d * EDIM * EDIM, wo + (size_t)d * EDIM * EDIM,
        w1 + (size_t)d * EDIM * FFN,  w2 + (size_t)d * FFN * EDIM, wt);

    const float* bq_d = bq + (size_t)d * EDIM;
    const float* bk_d = bk + (size_t)d * EDIM;
    const float* bv_d = bv + (size_t)d * EDIM;
    const float* bo_d = bo + (size_t)d * EDIM;
    const float* b1_d = b1 + (size_t)d * FFN;
    const float* b2_d = b2 + (size_t)d * EDIM;

    // QKV projections (V written transposed for the ctx matmul)
    gemm_kernel<0, false, false><<<blkE, 256, 0, stream>>>(xh, wt + WT_WQ,
        bq_d, nullptr, nullptr, qh, NTOK, EDIM, EDIM);
    gemm_kernel<0, false, false><<<blkE, 256, 0, stream>>>(xh, wt + WT_WK,
        bk_d, nullptr, nullptr, kh, NTOK, EDIM, EDIM);
    gemm_kernel<0, false, true><<<blkE, 256, 0, stream>>>(xh, wt + WT_WV,
        bv_d, nullptr, nullptr, vT, NTOK, EDIM, EDIM);

    // causal attention
    attn_kernel<<<dim3(LSEQ / 16, NHEAD, 8), 32, 0, stream>>>(qh, kh, vT, ctxh);

    // output projection + residual
    gemm_kernel<0, true, false><<<blkE, 256, 0, stream>>>(ctxh, wt + WT_WO,
        bo_d, xf, xf, xh, NTOK, EDIM, EDIM);

    // FFN
    gemm_kernel<1, false, false><<<blkF1, 256, 0, stream>>>(xh, wt + WT_W1,
        b1_d, nullptr, nullptr, hbuf, NTOK, FFN, EDIM);
    gemm_kernel<0, true, false><<<blkE, 256, 0, stream>>>(hbuf, wt + WT_W2,
        b2_d, xf, xf, xh, NTOK, EDIM, FFN);
  }

  final_kernel<<<8, 256, 0, stream>>>(xf, wf, bf, s, out);
}
